// Attention_17351667875992
// MI455X (gfx1250) — matrix operand order repaired and checked by reading, not compiled, not run on hardware
//
#include <hip/hip_runtime.h>

// ---------------------------------------------------------------------------
// Types for CDNA5 WMMA (wave32): v_wmma_f32_16x16x32_f16
// ---------------------------------------------------------------------------
typedef _Float16 hv8  __attribute__((ext_vector_type(8)));
typedef _Float16 v16h __attribute__((ext_vector_type(16)));
typedef float    v8f  __attribute__((ext_vector_type(8)));

#define B_DIM   2
#define S_LEN   2048
#define D_MODEL 1024
#define N_HEADS 16
#define HEAD_D  64
#define ROWS    (B_DIM * S_LEN)   // 4096
#define N_QKV   (3 * D_MODEL)     // 3072

// ---- gfx1250 async global->LDS copy path (compile-safe guard) -------------
#if defined(__has_builtin)
#if __has_builtin(__builtin_amdgcn_global_load_async_to_lds_b128) && \
    __has_builtin(__builtin_amdgcn_s_wait_asynccnt)
#define USE_ASYNC_LDS 1
#endif
#endif
#ifndef USE_ASYNC_LDS
#define USE_ASYNC_LDS 0
#endif

#if USE_ASYNC_LDS
typedef int v4i __attribute__((ext_vector_type(4)));
typedef __attribute__((address_space(1))) v4i* as1_v4p;
typedef __attribute__((address_space(3))) v4i* as3_v4p;
__device__ __forceinline__ void cp16_async(const void* g, void* l) {
  __builtin_amdgcn_global_load_async_to_lds_b128((as1_v4p)(void*)g,
                                                 (as3_v4p)l, 0, 0);
}
__device__ __forceinline__ void async_wait0() {
  __builtin_amdgcn_s_wait_asynccnt(0);
}
#else
__device__ __forceinline__ void cp16_async(const void* g, void* l) {
  *(hv8*)l = *(const hv8*)g;
}
__device__ __forceinline__ void async_wait0() {}
#endif

__device__ __forceinline__ v8f v8f_zero() {
  v8f z = {0.f, 0.f, 0.f, 0.f, 0.f, 0.f, 0.f, 0.f};
  return z;
}

// A-operand fragment (16x32 f16): lane half h holds K = h*16+0..15
__device__ __forceinline__ v16h frag_a16(const _Float16* row, int half, int k0) {
  hv8 lo = *(const hv8*)(row + k0 + half * 16);
  hv8 hi = *(const hv8*)(row + k0 + half * 16 + 8);
  v16h r;
#pragma unroll
  for (int i = 0; i < 8; ++i) { r[i] = lo[i]; r[i + 8] = hi[i]; }
  return r;
}

// B-operand fragment (32x16 f16): lane half h holds K = h*16+0..15 of its column.
__device__ __forceinline__ v16h frag_b16(const _Float16* col, int half, int k0) {
  hv8 lo = *(const hv8*)(col + k0 + half * 16);
  hv8 hi = *(const hv8*)(col + k0 + half * 16 + 8);
  v16h r;
#pragma unroll
  for (int i = 0; i < 8; ++i) { r[i] = lo[i]; r[i + 8] = hi[i]; }
  return r;
}

__device__ __forceinline__ v8f wmma_f16(v16h a, v16h b, v8f c) {
  return __builtin_amdgcn_wmma_f32_16x16x32_f16(false, a, false, b, (short)0, c,
                                                false, false);
}

__device__ __forceinline__ unsigned pack2h(_Float16 a, _Float16 b) {
  union { _Float16 h2[2]; unsigned u; } cv;
  cv.h2[0] = a; cv.h2[1] = b;
  return cv.u;
}

// ---------------------------------------------------------------------------
// Kernel 1: LayerNorm (f32 in) -> f16 normalized activations [ROWS][D_MODEL]
// ---------------------------------------------------------------------------
__global__ __launch_bounds__(256) void ln_kernel(const float* __restrict__ x,
                                                 const float* __restrict__ g,
                                                 const float* __restrict__ b,
                                                 _Float16* __restrict__ xh) {
  const int row = blockIdx.x;
  const int t = threadIdx.x;
  const float* xr = x + (size_t)row * D_MODEL;
  const float4 xv = ((const float4*)xr)[t];
  float s = xv.x + xv.y + xv.z + xv.w;
  float ss = xv.x * xv.x + xv.y * xv.y + xv.z * xv.z + xv.w * xv.w;
#pragma unroll
  for (int m = 1; m < 32; m <<= 1) {
    s += __shfl_xor(s, m, 32);
    ss += __shfl_xor(ss, m, 32);
  }
  __shared__ float red0[8], red1[8];
  const int wid = t >> 5, lane = t & 31;
  if (lane == 0) { red0[wid] = s; red1[wid] = ss; }
  __syncthreads();
  s = 0.f; ss = 0.f;
#pragma unroll
  for (int i = 0; i < 8; ++i) { s += red0[i]; ss += red1[i]; }
  const float mean = s * (1.f / D_MODEL);
  const float var = ss * (1.f / D_MODEL) - mean * mean;
  const float rstd = __builtin_amdgcn_rsqf(var + 1e-6f);
  const float4 gv = ((const float4*)g)[t];
  const float4 bv = ((const float4*)b)[t];
  union { _Float16 h[4]; unsigned long long u; } cv;
  cv.h[0] = (_Float16)((xv.x - mean) * rstd * gv.x + bv.x);
  cv.h[1] = (_Float16)((xv.y - mean) * rstd * gv.y + bv.y);
  cv.h[2] = (_Float16)((xv.z - mean) * rstd * gv.z + bv.z);
  cv.h[3] = (_Float16)((xv.w - mean) * rstd * gv.w + bv.w);
  ((unsigned long long*)(xh + (size_t)row * D_MODEL))[t] = cv.u;
}

// ---------------------------------------------------------------------------
// Kernel 2: f32 -> f16 cast (weights), 4-wide
// ---------------------------------------------------------------------------
__global__ __launch_bounds__(256) void cast_kernel(const float* __restrict__ src,
                                                   _Float16* __restrict__ dst,
                                                   int n4) {
  int i = blockIdx.x * 256 + threadIdx.x;
  const int stride = gridDim.x * 256;
  for (; i < n4; i += stride) {
    const float4 f = ((const float4*)src)[i];
    union { _Float16 h[4]; unsigned long long u; } cv;
    cv.h[0] = (_Float16)f.x;
    cv.h[1] = (_Float16)f.y;
    cv.h[2] = (_Float16)f.z;
    cv.h[3] = (_Float16)f.w;
    ((unsigned long long*)dst)[i] = cv.u;
  }
}

// ---------------------------------------------------------------------------
// Kernel 3: QKV GEMM  C[4096 x 3072] = xh[4096 x 1024] * w[1024 x 3072]
// Block tile 128x128, BK=64, 256 threads = 8 waves (4M x 2N), wave tile 32x64.
// ---------------------------------------------------------------------------
__global__ __launch_bounds__(256) void gemm_qkv_kernel(
    const _Float16* __restrict__ A, const _Float16* __restrict__ W,
    _Float16* __restrict__ qh, _Float16* __restrict__ kh,
    _Float16* __restrict__ vh) {
  __shared__ __align__(16) _Float16 lds_a[128][72];   // [M][K] (padded)
  __shared__ __align__(16) _Float16 lds_bt[128][72];  // [N][K] transposed

  const int t = threadIdx.x;
  const int lane = t & 31, wv = t >> 5;
  const int half = lane >> 4, l16 = lane & 15;
  const int mBase = blockIdx.y * 128, nBase = blockIdx.x * 128;
  const int wM = (wv & 3) * 32, wN = (wv >> 2) * 64;

  v8f acc[2][4];
#pragma unroll
  for (int i = 0; i < 2; ++i)
#pragma unroll
    for (int j = 0; j < 4; ++j) acc[i][j] = v8f_zero();

  const int arow = t >> 1, acol = (t & 1) * 32;  // A: 32 halves/thread
  const int kp = t >> 3, ncol = (t & 7) * 16;    // B: 2 K-rows x 16 N/thread

  for (int k0 = 0; k0 < D_MODEL; k0 += 64) {
    __syncthreads();
    {  // stage A [128][64] -- async 16B copies when available
      const _Float16* gp = A + (size_t)(mBase + arow) * D_MODEL + k0 + acol;
      cp16_async(gp, &lds_a[arow][acol]);
      cp16_async(gp + 8, &lds_a[arow][acol + 8]);
      cp16_async(gp + 16, &lds_a[arow][acol + 16]);
      cp16_async(gp + 24, &lds_a[arow][acol + 24]);
    }
    {  // stage B transposed with packed (k,k+1) dword stores
      const _Float16* g0 = W + (size_t)(k0 + 2 * kp) * N_QKV + nBase + ncol;
      const _Float16* g1 = g0 + N_QKV;
      hv8 a0 = *(const hv8*)(g0);
      hv8 a1 = *(const hv8*)(g0 + 8);
      hv8 b0 = *(const hv8*)(g1);
      hv8 b1 = *(const hv8*)(g1 + 8);
#pragma unroll
      for (int i = 0; i < 8; ++i) {
        *(unsigned*)&lds_bt[ncol + i][2 * kp] = pack2h(a0[i], b0[i]);
        *(unsigned*)&lds_bt[ncol + 8 + i][2 * kp] = pack2h(a1[i], b1[i]);
      }
    }
    if (k0 + 64 < D_MODEL) {  // prefetch next K-step while this one computes
      __builtin_prefetch(A + (size_t)(mBase + arow) * D_MODEL + k0 + 64 + acol, 0, 3);
      __builtin_prefetch(W + (size_t)(k0 + 64 + 2 * kp) * N_QKV + nBase + ncol, 0, 3);
    }
    async_wait0();
    __syncthreads();
#pragma unroll
    for (int ks = 0; ks < 64; ks += 32) {
      v16h af[2], bf[4];
#pragma unroll
      for (int mi = 0; mi < 2; ++mi)
        af[mi] = frag_a16(&lds_a[wM + mi * 16 + l16][0], half, ks);
#pragma unroll
      for (int ni = 0; ni < 4; ++ni)
        bf[ni] = frag_b16(&lds_bt[wN + ni * 16 + l16][0], half, ks);
#pragma unroll
      for (int mi = 0; mi < 2; ++mi)
#pragma unroll
        for (int ni = 0; ni < 4; ++ni)
          acc[mi][ni] = wmma_f16(af[mi], bf[ni], acc[mi][ni]);
    }
  }

  // Epilogue: C layout => lane holds column n = +l16, rows m = v + 8*half.
#pragma unroll
  for (int ni = 0; ni < 4; ++ni) {
    const int n = nBase + wN + ni * 16 + l16;
    const int which = n >> 10;  // 0=q 1=k 2=v
    const int hh = (n >> 6) & 15;
    const int dd = n & 63;
    _Float16* dp = (which == 0) ? qh : ((which == 1) ? kh : vh);
#pragma unroll
    for (int mi = 0; mi < 2; ++mi) {
#pragma unroll
      for (int v = 0; v < 8; ++v) {
        const int m = mBase + wM + mi * 16 + v + 8 * half;
        const int bb = m >> 11;
        const int srow = m & (S_LEN - 1);
        dp[(((size_t)(bb * N_HEADS + hh)) * S_LEN + srow) * HEAD_D + dd] =
            (_Float16)acc[mi][ni][v];
      }
    }
  }
}

// ---------------------------------------------------------------------------
// Kernel 4: flash attention.  Block = (64 queries) x (b,h); 128 thr = 4 waves.
// S^T = K*Q^T so softmax over keys is per-lane; O^T = V^T * P.
// ---------------------------------------------------------------------------
__global__ __launch_bounds__(128) void attn_kernel(
    const _Float16* __restrict__ qh, const _Float16* __restrict__ kh,
    const _Float16* __restrict__ vh, _Float16* __restrict__ oh) {
  __shared__ __align__(16) _Float16 lds_k[32][72];   // [kv][d]
  __shared__ __align__(16) _Float16 lds_vt[64][40];  // [d][kv]

  const int t = threadIdx.x;
  const int lane = t & 31, wv = t >> 5;
  const int half = lane >> 4, l16 = lane & 15;
  const int qtile = blockIdx.x, hh = blockIdx.y, bb = blockIdx.z;
  const size_t bh = (size_t)(bb * N_HEADS + hh);

  const _Float16* kbase = kh + bh * S_LEN * HEAD_D;
  const _Float16* vbase = vh + bh * S_LEN * HEAD_D;
  const int qrow = qtile * 64 + wv * 16 + l16;
  const _Float16* qptr = qh + (bh * S_LEN + qrow) * HEAD_D;

  v16h qf[2];
#pragma unroll
  for (int s = 0; s < 2; ++s) qf[s] = frag_b16(qptr, half, s * 32);

  v8f accO[4];
#pragma unroll
  for (int i = 0; i < 4; ++i) accO[i] = v8f_zero();
  float mrun = -1e30f, lrun = 0.f;
  const float c2 = 0.125f * 1.44269504088896340736f;  // scale * log2(e)

  const int grow = t >> 2, gcol = (t & 3) * 16;  // K stage: 32 x 64
  const int kp = t >> 3, dcol = (t & 7) * 8;     // V^T stage: 2 kv-rows x 8 d

  for (int c0 = 0; c0 < S_LEN; c0 += 32) {
    __syncthreads();
    {  // stage K chunk [32][64] -- async when available
      const _Float16* gp = kbase + (size_t)(c0 + grow) * HEAD_D + gcol;
      cp16_async(gp, &lds_k[grow][gcol]);
      cp16_async(gp + 8, &lds_k[grow][gcol + 8]);
    }
    {  // stage V^T [64][32] with packed (kv,kv+1) dword stores
      const _Float16* g0 = vbase + (size_t)(c0 + 2 * kp) * HEAD_D + dcol;
      const _Float16* g1 = g0 + HEAD_D;
      hv8 v0 = *(const hv8*)(g0);
      hv8 v1 = *(const hv8*)(g1);
#pragma unroll
      for (int i = 0; i < 8; ++i)
        *(unsigned*)&lds_vt[dcol + i][2 * kp] = pack2h(v0[i], v1[i]);
    }
    if (c0 + 32 < S_LEN) {  // prefetch next kv chunk
      __builtin_prefetch(kbase + (size_t)(c0 + 32 + grow) * HEAD_D + gcol, 0, 3);
      __builtin_prefetch(vbase + (size_t)(c0 + 32 + 2 * kp) * HEAD_D + dcol, 0, 3);
    }
    async_wait0();
    __syncthreads();

    // S^T tiles: M = kv (2 tiles of 16), N = q (lane), K-reduce over d = 64
    v8f accS[2] = {v8f_zero(), v8f_zero()};
#pragma unroll
    for (int s = 0; s < 2; ++s)
#pragma unroll
      for (int mt = 0; mt < 2; ++mt) {
        v16h af = frag_a16(&lds_k[mt * 16 + l16][0], half, s * 32);
        accS[mt] = wmma_f16(af, qf[s], accS[mt]);
      }

    // Online softmax (base-2). Lane + xor-16 partner cover all 32 kv.
    float tv[16];
    float cmax = -1e30f;
#pragma unroll
    for (int mt = 0; mt < 2; ++mt)
#pragma unroll
      for (int v = 0; v < 8; ++v) {
        const float s2 = accS[mt][v] * c2;
        tv[mt * 8 + v] = s2;
        cmax = fmaxf(cmax, s2);
      }
    cmax = fmaxf(cmax, __shfl_xor(cmax, 16, 32));
    const float mnew = fmaxf(mrun, cmax);
    const float sf = __builtin_amdgcn_exp2f(mrun - mnew);
    float p[16], psum = 0.f;
#pragma unroll
    for (int i = 0; i < 16; ++i) {
      p[i] = __builtin_amdgcn_exp2f(tv[i] - mnew);
      psum += p[i];
    }
    psum += __shfl_xor(psum, 16, 32);
    lrun = lrun * sf + psum;
    mrun = mnew;
#pragma unroll
    for (int i = 0; i < 4; ++i)
#pragma unroll
      for (int v = 0; v < 8; ++v) accO[i][v] *= sf;

    // Build P as WMMA B-operand (lane half h holds kv = h*16+0..15):
    // f16 packs + xor-16 shuffles, no LDS round-trip.
    unsigned po0[4], po1[4];
#pragma unroll
    for (int i = 0; i < 4; ++i) {
      po0[i] = pack2h((_Float16)p[2 * i], (_Float16)p[2 * i + 1]);
      po1[i] = pack2h((_Float16)p[8 + 2 * i], (_Float16)p[8 + 2 * i + 1]);
    }
    unsigned ot0[4], ot1[4];
#pragma unroll
    for (int i = 0; i < 4; ++i) {
      ot0[i] = (unsigned)__shfl_xor((int)po0[i], 16, 32);
      ot1[i] = (unsigned)__shfl_xor((int)po1[i], 16, 32);
    }
    union { unsigned u[8]; v16h v; } pf;
#pragma unroll
    for (int i = 0; i < 4; ++i) {
      pf.u[i] = half ? ot1[i] : po0[i];
      pf.u[4 + i] = half ? po1[i] : ot0[i];
    }

    // O^T += V^T * P  (M = d, 4 tiles of 16; K = 32 kv, one WMMA step)
#pragma unroll
    for (int mt = 0; mt < 4; ++mt) {
      v16h af = frag_a16(&lds_vt[mt * 16 + l16][0], half, 0);
      accO[mt] = wmma_f16(af, pf.v, accO[mt]);
    }
  }

  // Normalize and store: out[b][qrow][hh*64 + d], d = mt*16 + 8*half + v.
  const float invl = 1.f / lrun;
  _Float16* orow = oh + ((size_t)bb * S_LEN + qrow) * D_MODEL + hh * HEAD_D;
#pragma unroll
  for (int mt = 0; mt < 4; ++mt) {
#pragma unroll
    for (int v = 0; v < 8; v += 2) {
      const int d = mt * 16 + 8 * half + v;
      *(unsigned*)(&orow[d]) = pack2h((_Float16)(accO[mt][v] * invl),
                                      (_Float16)(accO[mt][v + 1] * invl));
    }
  }
}

// ---------------------------------------------------------------------------
// Kernel 5: output projection  out[4096x1024] = attn[4096x1024]*W + bias (f32)
// ---------------------------------------------------------------------------
__global__ __launch_bounds__(256) void gemm_out_kernel(
    const _Float16* __restrict__ A, const _Float16* __restrict__ W,
    const float* __restrict__ bias, float* __restrict__ out) {
  __shared__ __align__(16) _Float16 lds_a[128][72];
  __shared__ __align__(16) _Float16 lds_bt[128][72];

  const int t = threadIdx.x;
  const int lane = t & 31, wv = t >> 5;
  const int half = lane >> 4, l16 = lane & 15;
  const int mBase = blockIdx.y * 128, nBase = blockIdx.x * 128;
  const int wM = (wv & 3) * 32, wN = (wv >> 2) * 64;

  v8f acc[2][4];
#pragma unroll
  for (int i = 0; i < 2; ++i)
#pragma unroll
    for (int j = 0; j < 4; ++j) acc[i][j] = v8f_zero();

  const int arow = t >> 1, acol = (t & 1) * 32;
  const int kp = t >> 3, ncol = (t & 7) * 16;

  for (int k0 = 0; k0 < D_MODEL; k0 += 64) {
    __syncthreads();
    {
      const _Float16* gp = A + (size_t)(mBase + arow) * D_MODEL + k0 + acol;
      cp16_async(gp, &lds_a[arow][acol]);
      cp16_async(gp + 8, &lds_a[arow][acol + 8]);
      cp16_async(gp + 16, &lds_a[arow][acol + 16]);
      cp16_async(gp + 24, &lds_a[arow][acol + 24]);
    }
    {
      const _Float16* g0 = W + (size_t)(k0 + 2 * kp) * D_MODEL + nBase + ncol;
      const _Float16* g1 = g0 + D_MODEL;
      hv8 a0 = *(const hv8*)(g0);
      hv8 a1 = *(const hv8*)(g0 + 8);
      hv8 b0 = *(const hv8*)(g1);
      hv8 b1 = *(const hv8*)(g1 + 8);
#pragma unroll
      for (int i = 0; i < 8; ++i) {
        *(unsigned*)&lds_bt[ncol + i][2 * kp] = pack2h(a0[i], b0[i]);
        *(unsigned*)&lds_bt[ncol + 8 + i][2 * kp] = pack2h(a1[i], b1[i]);
      }
    }
    if (k0 + 64 < D_MODEL) {
      __builtin_prefetch(A + (size_t)(mBase + arow) * D_MODEL + k0 + 64 + acol, 0, 3);
      __builtin_prefetch(W + (size_t)(k0 + 64 + 2 * kp) * D_MODEL + nBase + ncol, 0, 3);
    }
    async_wait0();
    __syncthreads();
#pragma unroll
    for (int ks = 0; ks < 64; ks += 32) {
      v16h af[2], bf[4];
#pragma unroll
      for (int mi = 0; mi < 2; ++mi)
        af[mi] = frag_a16(&lds_a[wM + mi * 16 + l16][0], half, ks);
#pragma unroll
      for (int ni = 0; ni < 4; ++ni)
        bf[ni] = frag_b16(&lds_bt[wN + ni * 16 + l16][0], half, ks);
#pragma unroll
      for (int mi = 0; mi < 2; ++mi)
#pragma unroll
        for (int ni = 0; ni < 4; ++ni)
          acc[mi][ni] = wmma_f16(af[mi], bf[ni], acc[mi][ni]);
    }
  }

#pragma unroll
  for (int ni = 0; ni < 4; ++ni) {
    const int n = nBase + wN + ni * 16 + l16;
    const float bn = bias[n];
#pragma unroll
    for (int mi = 0; mi < 2; ++mi) {
#pragma unroll
      for (int v = 0; v < 8; ++v) {
        const int m = mBase + wM + mi * 16 + v + 8 * half;
        out[(size_t)m * D_MODEL + n] = acc[mi][ni][v] + bn;
      }
    }
  }
}

// ---------------------------------------------------------------------------
extern "C" void kernel_launch(void* const* d_in, const int* in_sizes, int n_in,
                              void* d_out, int out_size, void* d_ws,
                              size_t ws_size, hipStream_t stream) {
  (void)in_sizes; (void)n_in; (void)out_size; (void)ws_size;
  const float* x = (const float*)d_in[0];
  const float* ln_scale = (const float*)d_in[1];
  const float* ln_bias = (const float*)d_in[2];
  const float* w_qkv = (const float*)d_in[3];
  const float* w_out = (const float*)d_in[4];
  const float* b_out = (const float*)d_in[5];
  float* out = (float*)d_out;

  size_t off = 0;
  char* base = (char*)d_ws;
  auto alloc = [&](size_t bytes) -> char* {
    char* p = base + off;
    off += (bytes + 255) & ~(size_t)255;
    return p;
  };
  _Float16* xh = (_Float16*)alloc((size_t)ROWS * D_MODEL * 2);      // 8 MB
  _Float16* wqkvh = (_Float16*)alloc((size_t)D_MODEL * N_QKV * 2);  // 6 MB
  _Float16* woh = (_Float16*)alloc((size_t)D_MODEL * D_MODEL * 2);  // 2 MB
  _Float16* qhp = (_Float16*)alloc((size_t)ROWS * D_MODEL * 2);     // 8 MB
  _Float16* khp = (_Float16*)alloc((size_t)ROWS * D_MODEL * 2);     // 8 MB
  _Float16* vhp = (_Float16*)alloc((size_t)ROWS * D_MODEL * 2);     // 8 MB
  _Float16* attnh = xh;  // xh dead after QKV GEMM; reuse for attention output

  ln_kernel<<<ROWS, 256, 0, stream>>>(x, ln_scale, ln_bias, xh);
  cast_kernel<<<1024, 256, 0, stream>>>(w_qkv, wqkvh, (D_MODEL * N_QKV) / 4);
  cast_kernel<<<512, 256, 0, stream>>>(w_out, woh, (D_MODEL * D_MODEL) / 4);
  gemm_qkv_kernel<<<dim3(N_QKV / 128, ROWS / 128), 256, 0, stream>>>(
      xh, wqkvh, qhp, khp, vhp);
  attn_kernel<<<dim3(S_LEN / 64, N_HEADS, B_DIM), 128, 0, stream>>>(
      qhp, khp, vhp, attnh);
  gemm_out_kernel<<<dim3(D_MODEL / 128, ROWS / 128), 256, 0, stream>>>(
      attnh, woh, b_out, out);
}
